// TransformerBlock_86595130622136
// MI455X (gfx1250) — matrix operand order repaired and checked by reading, not compiled, not run on hardware
//
#include <hip/hip_runtime.h>
#include <hip/hip_bf16.h>

// ---------------------------------------------------------------------------
// Transformer block for MI455X (gfx1250, wave32, WMMA + TDM).
// All matmuls: V_WMMA_F32_16X16X32_BF16 (bf16 in, f32 accumulate).
// GEMM A-tiles are DMA'd into LDS by the Tensor Data Mover (tensor_load_to_lds,
// double-buffered, tracked with TENSORcnt); weight B-fragments are contiguous
// 32B global loads from pre-transposed [N][K] weights (L2-resident).
// ---------------------------------------------------------------------------

typedef __bf16 bf16;
typedef __attribute__((ext_vector_type(8)))  bf16  v8bf;
typedef __attribute__((ext_vector_type(16))) bf16  v16bf;
typedef __attribute__((ext_vector_type(8)))  float v8f;
typedef unsigned int u32x4 __attribute__((ext_vector_type(4)));
typedef int          i32x4 __attribute__((ext_vector_type(4)));
typedef int          i32x8 __attribute__((ext_vector_type(8)));

#define B_   4
#define S_   2048
#define D_   1024
#define H_   16
#define HD_  64
#define F_   4096
#define EPS_ 1e-5f
#define MROWS (B_ * S_) /* 8192 */

static __device__ inline v16bf mk16(v8bf lo, v8bf hi) {
  return __builtin_shufflevector(lo, hi, 0,1,2,3,4,5,6,7,8,9,10,11,12,13,14,15);
}

static __device__ inline v8f wmma_bf16(v16bf a, v16bf b, v8f c) {
  return __builtin_amdgcn_wmma_f32_16x16x32_bf16(
      false, a, false, b, (short)0, c, false, false);
}

// TDM: DMA a 128x32 bf16 tile (row stride K elems) from global into LDS.
// Pad: after every 16 DWORDs (one 32-elem row) insert 4 DWORDs -> LDS row
// stride 40 bf16, matching the conflict-free As layout.
static __device__ inline void tdm_load_tile(const bf16* gsrc, unsigned lds_off,
                                            int K) {
  unsigned long long ga = (unsigned long long)(size_t)gsrc;
  u32x4 g0;
  g0[0] = 1u;                                   // count=1 (valid user D#)
  g0[1] = lds_off;                              // lds_addr (bytes)
  g0[2] = (unsigned)(ga & 0xFFFFFFFFu);         // global_addr[31:0]
  g0[3] = (unsigned)((ga >> 32) & 0x1FFFFFFu)   // global_addr[56:32]
          | (2u << 30);                         // type=2 ("image")

  i32x8 g1;
  g1[0] = (1 << 16)      // data_size = 2 bytes
        | (1 << 20)      // pad_enable
        | (3 << 22)      // pad_interval: 16 DWORDs
        | (3 << 25);     // pad_amount: 4 DWORDs
  g1[1] = (int)(32u << 16);          // tensor_dim0[15:0] = 32 (bits 63:48)
  g1[2] = (int)(128u << 16);         // dim0 hi=0; tensor_dim1 lo = 128
  g1[3] = (int)(32u << 16);          // dim1 hi=0; tile_dim0 = 32
  g1[4] = 128;                       // tile_dim1 = 128; tile_dim2 = 0
  g1[5] = K;                         // tensor_dim0_stride[31:0] = K elems
  g1[6] = 0;                         // stride0 hi; stride1 lo
  g1[7] = 0;                         // stride1 hi

  i32x4 z4 = {0, 0, 0, 0};
#if __clang_major__ >= 23
  i32x8 z8 = {0, 0, 0, 0, 0, 0, 0, 0};
  __builtin_amdgcn_tensor_load_to_lds(g0, g1, z4, z4, z8, 0);
#else
  __builtin_amdgcn_tensor_load_to_lds(g0, g1, z4, z4, 0);
#endif
}

// ------------------------------- casts / packing ---------------------------

__global__ void k_cvt(const float* __restrict__ s, bf16* __restrict__ d, int n) {
  int i = blockIdx.x * 256 + threadIdx.x;
  if (i < n) d[i] = (bf16)s[i];
}

// Transpose f32 [K][N] -> bf16 [N][K] (one-time, bandwidth-bound)
__global__ void k_cvt_t(const float* __restrict__ s, bf16* __restrict__ d,
                        int K, int N) {
  int i = blockIdx.x * 256 + threadIdx.x;
  if (i < K * N) {
    int k = i / N, n = i % N;
    d[(size_t)n * K + k] = (bf16)s[i];
  }
}

// W[h][d][e] (f32) -> Wt[(coloff + h*64 + e)][d] (bf16), Wt is [3072][1024]
__global__ void k_pack_w(const float* __restrict__ W, bf16* __restrict__ dst,
                         int coloff) {
  int i = blockIdx.x * 256 + threadIdx.x;
  if (i < H_ * D_ * HD_) {
    int h = i / (D_ * HD_);
    int rem = i % (D_ * HD_);
    int d = rem / HD_;
    int e = rem % HD_;
    dst[(size_t)(coloff + h * HD_ + e) * D_ + d] = (bf16)W[i];
  }
}

__global__ void k_pack_b(const float* __restrict__ bq, const float* __restrict__ bk,
                         const float* __restrict__ bv, float* __restrict__ dst) {
  int i = blockIdx.x * 256 + threadIdx.x;
  if (i < 3 * D_) {
    const float* s = (i < D_) ? bq : (i < 2 * D_) ? bk : bv;
    dst[i] = s[i & (D_ - 1)];
  }
}

// ------------------------------- GEMM --------------------------------------
// C[M,N] = A[M,K](bf16,row-major) * Bt[N][K](bf16, pre-transposed) + bias.
// 128x128 block tile, 8 waves (4M x 2N), each wave 32x64 = 8 wmma accums.
// A tiles: TDM -> LDS, double buffered; B frags: contiguous global loads.
template <bool RELU, bool OUTF, bool OUTB>
__global__ __launch_bounds__(256) void k_gemm(
    const bf16* __restrict__ A, const bf16* __restrict__ Bt,
    const float* __restrict__ bias, float* __restrict__ Cf,
    bf16* __restrict__ Cb, int M, int N, int K) {
  __shared__ bf16 As[2][128][40];

  const int t = threadIdx.x;
  const int lane = t & 31, w = t >> 5;
  const int half = lane >> 4, l15 = lane & 15;
  const int wm = w >> 1, wn = w & 1;
  const int m0 = blockIdx.y * 128, n0 = blockIdx.x * 128;
  const int nkt = K / 32;

  const bf16* bp[4];
#pragma unroll
  for (int ns = 0; ns < 4; ns++)
    bp[ns] = Bt + (size_t)(n0 + wn * 64 + ns * 16 + l15) * K + half * 16;

  v8f acc[2][4];
#pragma unroll
  for (int i = 0; i < 2; i++)
#pragma unroll
    for (int j = 0; j < 4; j++)
#pragma unroll
      for (int r = 0; r < 8; r++) acc[i][j][r] = 0.f;

  const unsigned lds_base = (unsigned)(size_t)&As[0][0][0];
  const unsigned lds_buf = (unsigned)sizeof(As[0]);
  const bf16* asrc = A + (size_t)m0 * K;

  if (w == 0) tdm_load_tile(asrc, lds_base, K);  // prologue: tile 0 -> buf 0

  for (int kt = 0; kt < nkt; kt++) {
    const int cur = kt & 1;
    if (w == 0) {
      if (kt + 1 < nkt) {  // issue next tile into the other buffer
        tdm_load_tile(asrc + (kt + 1) * 32, lds_base + (cur ^ 1) * lds_buf, K);
        __builtin_amdgcn_s_wait_tensorcnt(1);  // in-order: tile kt done
      } else {
        __builtin_amdgcn_s_wait_tensorcnt(0);
      }
    }
    __syncthreads();  // publish LDS tile to all waves

    const int k0 = kt * 32;
    v16bf a[2], b[4];
#pragma unroll
    for (int ms = 0; ms < 2; ms++) {
      int row = wm * 32 + ms * 16 + l15;  // A-layout: contiguous 16 along K
      a[ms] = mk16(*(const v8bf*)&As[cur][row][half * 16],
                   *(const v8bf*)&As[cur][row][half * 16 + 8]);
    }
#pragma unroll
    for (int ns = 0; ns < 4; ns++) {  // B-layout: contiguous 16 along K
      const bf16* p = bp[ns] + k0;
      b[ns] = mk16(*(const v8bf*)p, *(const v8bf*)(p + 8));
      if (k0 + 32 < K) __builtin_prefetch(p + 32, 0, 3);  // global_prefetch
    }
#pragma unroll
    for (int ms = 0; ms < 2; ms++)
#pragma unroll
      for (int ns = 0; ns < 4; ns++)
        acc[ms][ns] = wmma_bf16(a[ms], b[ns], acc[ms][ns]);
    __syncthreads();  // all reads done before buffer is overwritten
  }

#pragma unroll
  for (int ms = 0; ms < 2; ms++) {
#pragma unroll
    for (int ns = 0; ns < 4; ns++) {
      int col = n0 + wn * 64 + ns * 16 + l15;
      float bvl = bias[col];
#pragma unroll
      for (int r = 0; r < 8; r++) {
        int row = m0 + wm * 32 + ms * 16 + half * 8 + r;  // C-layout
        float v = acc[ms][ns][r] + bvl;
        if (RELU) v = fmaxf(v, 0.f);
        size_t idx = (size_t)row * N + col;
        if (OUTF) Cf[idx] = v;
        if (OUTB) Cb[idx] = (bf16)v;
      }
    }
  }
}

// ------------------------------- attention ---------------------------------
// Flash-style. Block: 128 query rows x one (b,h); 8 waves, wave = 16 rows.
// qkv: [8192][3072] bf16 (Q|K|V each 1024 cols, head h at h*64).
__global__ __launch_bounds__(256) void k_attn(const bf16* __restrict__ qkv,
                                              bf16* __restrict__ ctx) {
  __shared__ bf16 Vt[64][72];      // V tile transposed: [e][key], pad 72
  __shared__ bf16 Ps[8][16][72];   // per-wave P tile [row][key], pad 72

  const int t = threadIdx.x, lane = t & 31, w = t >> 5;
  const int half = lane >> 4, l15 = lane & 15;
  const int h = blockIdx.y, b = blockIdx.z;
  const int q0 = blockIdx.x * 128;
  const size_t rowbase = (size_t)b * S_;
  const int ldq = 3 * D_;

  v16bf qf[2];
  {
    const bf16* qp = qkv + (rowbase + q0 + w * 16 + l15) * ldq + h * HD_;
#pragma unroll
    for (int ks = 0; ks < 2; ks++) {
      int e0 = ks * 32 + half * 16;
      qf[ks] = mk16(*(const v8bf*)(qp + e0), *(const v8bf*)(qp + e0 + 8));
    }
  }

  float m_[8], l_[8];
  v8f o_[4];
#pragma unroll
  for (int r = 0; r < 8; r++) { m_[r] = -1e30f; l_[r] = 0.f; }
#pragma unroll
  for (int nt = 0; nt < 4; nt++)
#pragma unroll
    for (int r = 0; r < 8; r++) o_[nt][r] = 0.f;

  for (int kt = 0; kt < S_ / 64; kt++) {
    const int k0 = kt * 64;
    {  // cooperative V tile load, transposed into Vt[e][key]
      int key = t >> 2, e0 = (t & 3) * 16;
      const bf16* vp = qkv + (rowbase + k0 + key) * ldq + 2 * D_ + h * HD_ + e0;
      v8bf lo = *(const v8bf*)vp, hi = *(const v8bf*)(vp + 8);
#pragma unroll
      for (int i = 0; i < 8; i++) {
        Vt[e0 + i][key] = lo[i];
        Vt[e0 + 8 + i][key] = hi[i];
      }
    }
    __syncthreads();

    // scores S = Q * K^T  (K^T B-frags straight from global: contiguous in e)
    v8f s[4];
#pragma unroll
    for (int nt = 0; nt < 4; nt++)
#pragma unroll
      for (int r = 0; r < 8; r++) s[nt][r] = 0.f;
#pragma unroll
    for (int ks = 0; ks < 2; ks++) {
#pragma unroll
      for (int nt = 0; nt < 4; nt++) {
        const bf16* kp = qkv + (rowbase + k0 + nt * 16 + l15) * ldq + D_ +
                         h * HD_ + ks * 32 + half * 16;
        v16bf kf = mk16(*(const v8bf*)kp, *(const v8bf*)(kp + 8));
        s[nt] = wmma_bf16(qf[ks], kf, s[nt]);
      }
    }

    // online softmax (rows live in 16-lane halves; width-16 shfl reductions)
#pragma unroll
    for (int r = 0; r < 8; r++) {
      float tm = -1e30f;
#pragma unroll
      for (int nt = 0; nt < 4; nt++) {
        s[nt][r] *= 0.125f;  // 1/sqrt(HD)
        tm = fmaxf(tm, s[nt][r]);
      }
#pragma unroll
      for (int off = 8; off >= 1; off >>= 1) tm = fmaxf(tm, __shfl_xor(tm, off, 16));
      float mn = fmaxf(m_[r], tm);
      float alpha = __expf(m_[r] - mn);
      m_[r] = mn;
      l_[r] *= alpha;
#pragma unroll
      for (int nt = 0; nt < 4; nt++) o_[nt][r] *= alpha;
      float rs = 0.f;
      int rl = half * 8 + r;
#pragma unroll
      for (int nt = 0; nt < 4; nt++) {
        float p = __expf(s[nt][r] - mn);
        rs += p;
        Ps[w][rl][nt * 16 + l15] = (bf16)p;  // C-layout -> LDS (A-layout src)
      }
#pragma unroll
      for (int off = 8; off >= 1; off >>= 1) rs += __shfl_xor(rs, off, 16);
      l_[r] += rs;
    }

    // O += P * V
#pragma unroll
    for (int ks2 = 0; ks2 < 2; ks2++) {
      int kk = ks2 * 32 + half * 16;
      v16bf pa = mk16(*(const v8bf*)&Ps[w][l15][kk],
                      *(const v8bf*)&Ps[w][l15][kk + 8]);
#pragma unroll
      for (int nt2 = 0; nt2 < 4; nt2++) {
        int e = nt2 * 16 + l15, key0 = ks2 * 32 + half * 16;
        v16bf vf = mk16(*(const v8bf*)&Vt[e][key0],
                        *(const v8bf*)&Vt[e][key0 + 8]);
        o_[nt2] = wmma_bf16(pa, vf, o_[nt2]);
      }
    }
    __syncthreads();
  }

  // write concat ctx (bf16, feeds out-proj GEMM)
#pragma unroll
  for (int nt2 = 0; nt2 < 4; nt2++) {
    int col = h * HD_ + nt2 * 16 + l15;
#pragma unroll
    for (int r = 0; r < 8; r++) {
      int row = q0 + w * 16 + half * 8 + r;
      ctx[(rowbase + row) * (size_t)(H_ * HD_) + col] = (bf16)(o_[nt2][r] / l_[r]);
    }
  }
}

// ------------------------------- residual + layernorm ----------------------
__global__ __launch_bounds__(256) void k_add_ln(
    const float* __restrict__ a, const float* __restrict__ bsrc,
    const float* __restrict__ g, const float* __restrict__ be,
    float* __restrict__ outf, bf16* __restrict__ outb) {
  __shared__ float red[256], red2[256];
  const int row = blockIdx.x, t = threadIdx.x;
  const float* pa = a + (size_t)row * D_;
  const float* pb = bsrc + (size_t)row * D_;
  float v[4];
  float s = 0.f, s2 = 0.f;
#pragma unroll
  for (int i = 0; i < 4; i++) {
    int c = t + i * 256;
    v[i] = pa[c] + pb[c];
    s += v[i];
    s2 += v[i] * v[i];
  }
  red[t] = s;
  red2[t] = s2;
  __syncthreads();
  for (int off = 128; off > 0; off >>= 1) {
    if (t < off) {
      red[t] += red[t + off];
      red2[t] += red2[t + off];
    }
    __syncthreads();
  }
  float mu = red[0] * (1.f / D_);
  float var = red2[0] * (1.f / D_) - mu * mu;
  float inv = rsqrtf(var + EPS_);
#pragma unroll
  for (int i = 0; i < 4; i++) {
    int c = t + i * 256;
    float y = (v[i] - mu) * inv * g[c] + be[c];
    if (outf) outf[(size_t)row * D_ + c] = y;
    if (outb) outb[(size_t)row * D_ + c] = (bf16)y;
  }
}

// ------------------------------- launch ------------------------------------
extern "C" void kernel_launch(void* const* d_in, const int* in_sizes, int n_in,
                              void* d_out, int out_size, void* d_ws, size_t ws_size,
                              hipStream_t stream) {
  const float* x   = (const float*)d_in[0];
  const float* Wq  = (const float*)d_in[1];
  const float* bq  = (const float*)d_in[2];
  const float* Wk  = (const float*)d_in[3];
  const float* bk  = (const float*)d_in[4];
  const float* Wv  = (const float*)d_in[5];
  const float* bv  = (const float*)d_in[6];
  const float* Wo  = (const float*)d_in[7];
  const float* bo  = (const float*)d_in[8];
  const float* W1  = (const float*)d_in[9];
  const float* b1  = (const float*)d_in[10];
  const float* W2  = (const float*)d_in[11];
  const float* b2  = (const float*)d_in[12];
  const float* g1  = (const float*)d_in[13];
  const float* be1 = (const float*)d_in[14];
  const float* g2  = (const float*)d_in[15];
  const float* be2 = (const float*)d_in[16];
  float* out = (float*)d_out;

  char* p = (char*)d_ws;
  auto alloc = [&](size_t bytes) {
    char* r = p;
    p += (bytes + 255) & ~(size_t)255;
    return r;
  };
  bf16*  xb    = (bf16*) alloc((size_t)MROWS * D_ * 2);
  bf16*  wqkvT = (bf16*) alloc((size_t)D_ * 3 * D_ * 2);  // [3072][1024]
  float* bqkv  = (float*)alloc((size_t)3 * D_ * 4);
  bf16*  woT   = (bf16*) alloc((size_t)D_ * D_ * 2);       // [1024][1024]
  bf16*  w1T   = (bf16*) alloc((size_t)D_ * F_ * 2);       // [4096][1024]
  bf16*  w2T   = (bf16*) alloc((size_t)F_ * D_ * 2);       // [1024][4096]
  bf16*  qkv   = (bf16*) alloc((size_t)MROWS * 3 * D_ * 2);
  bf16*  ctx   = (bf16*) alloc((size_t)MROWS * D_ * 2);
  float* tmpf  = (float*)alloc((size_t)MROWS * D_ * 4);  // attn_out, then ffn2
  float* x1f   = (float*)alloc((size_t)MROWS * D_ * 4);
  bf16*  x1b   = (bf16*) alloc((size_t)MROWS * D_ * 2);
  bf16*  ffn1  = (bf16*) alloc((size_t)MROWS * F_ * 2);

  dim3 blk(256);

  k_cvt<<<(MROWS * D_ + 255) / 256, blk, 0, stream>>>(x, xb, MROWS * D_);
  int nW = H_ * D_ * HD_;
  k_pack_w<<<(nW + 255) / 256, blk, 0, stream>>>(Wq, wqkvT, 0);
  k_pack_w<<<(nW + 255) / 256, blk, 0, stream>>>(Wk, wqkvT, D_);
  k_pack_w<<<(nW + 255) / 256, blk, 0, stream>>>(Wv, wqkvT, 2 * D_);
  k_pack_b<<<(3 * D_ + 255) / 256, blk, 0, stream>>>(bq, bk, bv, bqkv);
  k_cvt_t<<<(D_ * D_ + 255) / 256, blk, 0, stream>>>(Wo, woT, D_, D_);
  k_cvt_t<<<(D_ * F_ + 255) / 256, blk, 0, stream>>>(W1, w1T, D_, F_);
  k_cvt_t<<<(F_ * D_ + 255) / 256, blk, 0, stream>>>(W2, w2T, F_, D_);

  // QKV projection: [8192,1024] x [1024,3072] -> bf16
  k_gemm<false, false, true><<<dim3(3 * D_ / 128, MROWS / 128), blk, 0, stream>>>(
      xb, wqkvT, bqkv, nullptr, qkv, MROWS, 3 * D_, D_);

  // attention -> concat ctx bf16
  k_attn<<<dim3(S_ / 128, H_, B_), blk, 0, stream>>>(qkv, ctx);

  // out projection: [8192,1024] x [1024,1024] -> f32
  k_gemm<false, true, false><<<dim3(D_ / 128, MROWS / 128), blk, 0, stream>>>(
      ctx, woT, bo, tmpf, nullptr, MROWS, D_, D_);

  // x1 = LN(x + attn_out)
  k_add_ln<<<MROWS, blk, 0, stream>>>(x, tmpf, g1, be1, x1f, x1b);

  // FFN1 (+ReLU): [8192,1024] x [1024,4096] -> bf16
  k_gemm<true, false, true><<<dim3(F_ / 128, MROWS / 128), blk, 0, stream>>>(
      x1b, w1T, b1, nullptr, ffn1, MROWS, F_, D_);

  // FFN2: [8192,4096] x [4096,1024] -> f32 (reuse tmpf)
  k_gemm<false, true, false><<<dim3(D_ / 128, MROWS / 128), blk, 0, stream>>>(
      ffn1, w2T, b2, tmpf, nullptr, MROWS, D_, F_);

  // out = LN(x1 + ffn)
  k_add_ln<<<MROWS, blk, 0, stream>>>(x1f, tmpf, g2, be2, out, nullptr);
}
